// BaseNet_45569603011198
// MI455X (gfx1250) — matrix operand order repaired and checked by reading, not compiled, not run on hardware
//
#include <hip/hip_runtime.h>
#include <math.h>

// ---------------------------------------------------------------------------
// Types
// ---------------------------------------------------------------------------
typedef __attribute__((ext_vector_type(16))) _Float16 v16h;
typedef __attribute__((ext_vector_type(8)))  float    v8f;
typedef unsigned int uint32x4 __attribute__((ext_vector_type(4)));
typedef int          int32x4  __attribute__((ext_vector_type(4)));
typedef int          int32x8  __attribute__((ext_vector_type(8)));

#if defined(__has_builtin)
#if __has_builtin(__builtin_amdgcn_tensor_load_to_lds)
#define HAVE_TDM 1
#else
#define HAVE_TDM 0
#endif
#else
#define HAVE_TDM 0
#endif

// ---------------------------------------------------------------------------
// Device math helpers
// ---------------------------------------------------------------------------
__device__ __forceinline__ float gelu_f(float x)    { return 0.5f * x * (1.f + erff(x * 0.70710678118654752f)); }
__device__ __forceinline__ float sigmoid_f(float x) { return 1.f / (1.f + __expf(-x)); }
__device__ __forceinline__ float silu_f(float x)    { return x * sigmoid_f(x); }
__device__ __forceinline__ float softplus_f(float x){ return (x > 20.f) ? x : log1pf(__expf(x)); }

__device__ __forceinline__ float apply_act(float v, int ACT) {
  if (ACT == 1) return gelu_f(v);
  if (ACT == 2) return silu_f(v);
  if (ACT == 3) return sigmoid_f(v);
  if (ACT == 4) return softplus_f(v);
  return v;
}

// ---------------------------------------------------------------------------
// WMMA GEMM:  Y[M,N] = act( X[M,K](row stride lda) * W[N,K]^T + bias[N] )
// REQUIREMENTS (guaranteed by host code):
//   - M is a multiple of 32 (M = B*L everywhere)
//   - lda is a multiple of 4 and X points into workspace with >=1KB slack
//     after the logical end (tail loads over-read then mask with selects).
// Block = 128 threads (4 waves), C footprint 32(M) x 64(N).
// Each wave: two 16x16 accumulators sharing one B fragment
//   -> 2 x v_wmma_f32_16x16x32_f16 per K-step.
// W tile staged global->LDS via the Tensor Data Mover.
// ---------------------------------------------------------------------------
#define KT 32

template<int ACT>
__global__ __launch_bounds__(128) void k_gemm_wmma(
    const float* __restrict__ X, int lda,
    const float* __restrict__ W,
    const float* __restrict__ bias,
    float* __restrict__ Y, int ldy,
    int M, int N, int K)
{
  __shared__ float    Wst[64 * KT];
  __shared__ _Float16 As [32 * KT];
  __shared__ _Float16 Ws [64 * KT];

  const int tid  = threadIdx.x;
  const int lane = tid & 31;
  const int wv   = tid >> 5;
  const int m0   = blockIdx.y * 32;
  const int nb0  = blockIdx.x * 64;

  v8f acc0, acc1;
#pragma unroll
  for (int i = 0; i < 8; ++i) { acc0[i] = 0.f; acc1[i] = 0.f; }

  for (int kt = 0; kt < K; kt += KT) {
    __syncthreads();
    // ---- stage X tile (32 x KT) as f16: 256 float4 chunks, 2/thread ----
    if (kt + KT <= K) {
      // full tile: branch-free vector loads
#pragma unroll
      for (int c = tid; c < 32 * (KT / 4); c += 128) {
        int m  = c >> 3;
        int kq = (c & 7) << 2;
        const float* xp = X + (size_t)(m0 + m) * lda + kt + kq;
        float4 v = *(const float4*)xp;
        _Float16* ap = As + m * KT + kq;
        ap[0] = (_Float16)v.x; ap[1] = (_Float16)v.y;
        ap[2] = (_Float16)v.z; ap[3] = (_Float16)v.w;
      }
    } else {
      // K tail: over-read into workspace slack, mask with selects
#pragma unroll
      for (int c = tid; c < 32 * (KT / 4); c += 128) {
        int m  = c >> 3;
        int kq = (c & 7) << 2;
        int gk = kt + kq;
        const float* xp = X + (size_t)(m0 + m) * lda + gk;
        float4 v = *(const float4*)xp;
        v.x = (gk + 0 < K) ? v.x : 0.f;
        v.y = (gk + 1 < K) ? v.y : 0.f;
        v.z = (gk + 2 < K) ? v.z : 0.f;
        v.w = (gk + 3 < K) ? v.w : 0.f;
        _Float16* ap = As + m * KT + kq;
        ap[0] = (_Float16)v.x; ap[1] = (_Float16)v.y;
        ap[2] = (_Float16)v.z; ap[3] = (_Float16)v.w;
      }
    }
#if HAVE_TDM
    // ---- stage W tile (64 rows x KT) via Tensor Data Mover ----
    if (tid < 32) {
      unsigned long long ga = (unsigned long long)(const void*)(W + (size_t)nb0 * K + kt);
      unsigned int lds_addr = (unsigned int)(unsigned long long)(const void*)Wst; // low 32b = LDS offset
      unsigned int kRem = (unsigned int)(K - kt);          // tensor_dim0 (OOB reads -> 0)
      unsigned int rRem = (unsigned int)(N - nb0);
      if (rRem > 64u) rRem = 64u;                          // tensor_dim1
      uint32x4 g0 = { 1u,                                   // count=1, user descriptor
                      lds_addr,
                      (unsigned int)ga,
                      (unsigned int)((ga >> 32) & 0x1FFFFFFull) | (2u << 30) }; // type=2
      int32x8 g1 = { (int)(2u << 16),                                   // data_size = 4B
                     (int)((kRem & 0xFFFFu) << 16),                     // dim0 lo
                     (int)(((kRem >> 16) & 0xFFFFu) | ((rRem & 0xFFFFu) << 16)),
                     (int)(((rRem >> 16) & 0xFFFFu) | ((unsigned)KT << 16)),   // tile_dim0 = 32
                     (int)64,                                           // tile_dim1 = 64
                     (int)(unsigned int)K,                              // dim0 stride (elements)
                     0, 0 };
      int32x4 gz = { 0, 0, 0, 0 };
#if __clang_major__ >= 23
      int32x8 gz8 = { 0, 0, 0, 0, 0, 0, 0, 0 };
      __builtin_amdgcn_tensor_load_to_lds(g0, g1, gz, gz, gz8, 0);
#else
      __builtin_amdgcn_tensor_load_to_lds(g0, g1, gz, gz, 0);
#endif
      __builtin_amdgcn_s_wait_tensorcnt(0);
      if (kt + KT < K) __builtin_prefetch(W + (size_t)nb0 * K + kt + KT, 0, 1);
    }
    __syncthreads();
#pragma unroll
    for (int idx = tid; idx < 64 * KT; idx += 128) Ws[idx] = (_Float16)Wst[idx];
#else
    // fallback without TDM: guarded scalar loads (weights are harness inputs,
    // no over-read allowed)
    for (int c = tid; c < 64 * (KT / 4); c += 128) {
      int n  = c >> 3;
      int kq = (c & 7) << 2;
      int gn = nb0 + n, gk = kt + kq;
      float v0 = 0.f, v1 = 0.f, v2 = 0.f, v3 = 0.f;
      if (gn < N) {
        const float* wp = W + (size_t)gn * K + gk;
        if (gk + 0 < K) v0 = wp[0];
        if (gk + 1 < K) v1 = wp[1];
        if (gk + 2 < K) v2 = wp[2];
        if (gk + 3 < K) v3 = wp[3];
      }
      _Float16* wp16 = Ws + n * KT + kq;
      wp16[0] = (_Float16)v0; wp16[1] = (_Float16)v1;
      wp16[2] = (_Float16)v2; wp16[3] = (_Float16)v3;
    }
#endif
    __syncthreads();

    // ---- fragments per the 16x16x32 f16 VGPR layout ----
    const int ml = lane & 15, hi = lane >> 4;
    v16h a0, a1, bfr;
#pragma unroll
    for (int e = 0; e < 16; ++e) {
      int ka = (e & 7) + ((e >> 3) << 4) + (hi << 3);   // A: 16x32, K index
      a0[e]  = As[ml * KT + ka];
      a1[e]  = As[(16 + ml) * KT + ka];
      int kb = (e & 7) + ((e >> 3) << 4) + (hi << 3);   // B: 32x16, K index
      bfr[e] = Ws[(wv * 16 + ml) * KT + kb];
    }
    acc0 = __builtin_amdgcn_wmma_f32_16x16x32_f16(false, a0, false, bfr,
                                                  (short)0, acc0, false, false);
    acc1 = __builtin_amdgcn_wmma_f32_16x16x32_f16(false, a1, false, bfr,
                                                  (short)0, acc1, false, false);
  }

  // ---- epilogue: C layout m = r + 8*(lane>=16), n = lane&15 ----
  const int ml = lane & 15, hi = lane >> 4;
  const int gn = nb0 + wv * 16 + ml;
  if (gn < N) {
    float bv = bias ? bias[gn] : 0.f;
#pragma unroll
    for (int r = 0; r < 8; ++r) {
      int gm = m0 + r + (hi << 3);
      Y[(size_t)gm * ldy + gn]        = apply_act(acc0[r] + bv, ACT);
      Y[(size_t)(gm + 16) * ldy + gn] = apply_act(acc1[r] + bv, ACT);
    }
  }
}

// ---------------------------------------------------------------------------
// Mamba selective scan. dt,u,Y are (B,L,d_in); Bc/Cc are (B,L,*) views with
// row stride ldbc, 16B-aligned columns -> float4 loads. 8 lanes per channel,
// state h[N/8] in VGPRs, cross-lane C-contraction via shuffles.
// ---------------------------------------------------------------------------
template<int NPER>
__global__ __launch_bounds__(256) void k_scan(
    const float* __restrict__ dt, const float* __restrict__ Bc,
    const float* __restrict__ Cc, int ldbc,
    const float* __restrict__ u,  const float* __restrict__ A_log,
    const float* __restrict__ Dv, float* __restrict__ Y,
    int L, int d_in, int N)
{
  const int j    = threadIdx.x & 7;
  const int dloc = threadIdx.x >> 3;
  const int d    = blockIdx.x * 32 + dloc;
  const int b    = blockIdx.y;
  const int nbase = j * NPER;

  float a[NPER], h[NPER];
#pragma unroll
  for (int q = 0; q < NPER / 4; ++q) {
    float4 t = *(const float4*)(A_log + (size_t)d * N + nbase + q * 4);
    a[q*4+0] = -__expf(t.x); a[q*4+1] = -__expf(t.y);
    a[q*4+2] = -__expf(t.z); a[q*4+3] = -__expf(t.w);
  }
#pragma unroll
  for (int i = 0; i < NPER; ++i) h[i] = 0.f;
  const float dcoef = Dv[d];

  for (int t = 0; t < L; ++t) {
    size_t row = (size_t)b * L + t;
    float dtv = dt[row * d_in + d];
    float uv  = u [row * d_in + d];
    float du  = dtv * uv;
    const float* Bp = Bc + row * ldbc + nbase;
    const float* Cp = Cc + row * ldbc + nbase;
    float bv[NPER], cv[NPER];
#pragma unroll
    for (int q = 0; q < NPER / 4; ++q) {
      float4 tb = *(const float4*)(Bp + q * 4);
      float4 tc = *(const float4*)(Cp + q * 4);
      bv[q*4+0] = tb.x; bv[q*4+1] = tb.y; bv[q*4+2] = tb.z; bv[q*4+3] = tb.w;
      cv[q*4+0] = tc.x; cv[q*4+1] = tc.y; cv[q*4+2] = tc.z; cv[q*4+3] = tc.w;
    }
    float part = 0.f;
#pragma unroll
    for (int i = 0; i < NPER; ++i) {
      float hv = __expf(dtv * a[i]) * h[i] + du * bv[i];
      h[i] = hv;
      part += hv * cv[i];
    }
    part += __shfl_xor(part, 1, 32);
    part += __shfl_xor(part, 2, 32);
    part += __shfl_xor(part, 4, 32);
    if (j == 0) Y[row * d_in + d] = part + uv * dcoef;
  }
}

// ---------------------------------------------------------------------------
// Elementwise / small kernels
// ---------------------------------------------------------------------------
__global__ void k_layernorm(const float* __restrict__ X, const float* __restrict__ g,
                            const float* __restrict__ b, float* __restrict__ Y,
                            int M, int D)
{
  int row = blockIdx.x, lane = threadIdx.x;
  const float* xr = X + (size_t)row * D;
  float s = 0.f, ss = 0.f;
  for (int i = lane; i < D; i += 32) { float v = xr[i]; s += v; ss += v * v; }
  for (int o = 16; o; o >>= 1) { s += __shfl_xor(s, o, 32); ss += __shfl_xor(ss, o, 32); }
  float mu  = s / D;
  float var = ss / D - mu * mu;
  float inv = rsqrtf(var + 1e-5f);
  for (int i = lane; i < D; i += 32)
    Y[(size_t)row * D + i] = (xr[i] - mu) * inv * g[i] + b[i];
}

__global__ void k_conv1d_nch(const float* __restrict__ X, const float* __restrict__ Wt,
                             const float* __restrict__ bias, float* __restrict__ Y,
                             int B, int Cin, int Lin, int Cout, int Lout,
                             int Kw, int stride, int pad)
{
  size_t i = (size_t)blockIdx.x * 256 + threadIdx.x;
  size_t total = (size_t)B * Cout * Lout;
  if (i >= total) return;
  int t  = (int)(i % Lout);
  int co = (int)((i / Lout) % Cout);
  int b  = (int)(i / ((size_t)Lout * Cout));
  float acc = bias[co];
  int base = t * stride - pad;
  for (int ci = 0; ci < Cin; ++ci) {
    const float* xp = X + ((size_t)b * Cin + ci) * Lin;
    const float* wp = Wt + ((size_t)co * Cin + ci) * Kw;
    for (int k = 0; k < Kw; ++k) {
      int p = base + k;
      if (p >= 0 && p < Lin) acc += xp[p] * wp[k];
    }
  }
  Y[i] = acc;
}

__global__ void k_maxpool_nch(const float* __restrict__ X, float* __restrict__ Y,
                              int B, int C, int Lin, int k)
{
  int Lout = Lin / k;
  size_t i = (size_t)blockIdx.x * 256 + threadIdx.x;
  size_t total = (size_t)B * C * Lout;
  if (i >= total) return;
  int lo = (int)(i % Lout);
  size_t bc = i / Lout;
  const float* xp = X + bc * Lin + (size_t)lo * k;
  float m = xp[0];
  for (int kk = 1; kk < k; ++kk) m = fmaxf(m, xp[kk]);
  Y[i] = m;
}

__global__ void k_nch2nlc(const float* __restrict__ X, float* __restrict__ Y,
                          int B, int C, int L)
{
  size_t i = (size_t)blockIdx.x * 256 + threadIdx.x;
  size_t total = (size_t)B * C * L;
  if (i >= total) return;
  int c = (int)(i % C);
  int l = (int)((i / C) % L);
  int b = (int)(i / ((size_t)C * L));
  Y[i] = X[((size_t)b * C + c) * L + l];
}

__global__ void k_nlc2nch(const float* __restrict__ X, float* __restrict__ Y,
                          int B, int C, int L)
{
  size_t i = (size_t)blockIdx.x * 256 + threadIdx.x;
  size_t total = (size_t)B * C * L;
  if (i >= total) return;
  int l = (int)(i % L);
  int c = (int)((i / L) % C);
  int b = (int)(i / ((size_t)C * L));
  Y[i] = X[((size_t)b * L + l) * C + c];
}

__global__ void k_bn_gelu(const float* __restrict__ X, const float* __restrict__ g,
                          const float* __restrict__ be, const float* __restrict__ mn,
                          const float* __restrict__ vr, float* __restrict__ Y,
                          size_t total, int C)
{
  size_t i = (size_t)blockIdx.x * 256 + threadIdx.x;
  if (i >= total) return;
  int c = (int)(i % C);
  float sc = g[c] * rsqrtf(vr[c] + 1e-5f);
  Y[i] = gelu_f((X[i] - mn[c]) * sc + be[c]);
}

// depthwise conv along L, channel-last (B,L,C) layout
__global__ void k_dwconv_nlc(const float* __restrict__ X, const float* __restrict__ W,
                             const float* __restrict__ bias, float* __restrict__ Y,
                             int B, int L, int C, int Kw, int padl)
{
  size_t i = (size_t)blockIdx.x * 256 + threadIdx.x;
  size_t total = (size_t)B * L * C;
  if (i >= total) return;
  int c = (int)(i % C);
  int l = (int)((i / C) % L);
  int b = (int)(i / ((size_t)C * L));
  float acc = bias[c];
  for (int k = 0; k < Kw; ++k) {
    int p = l - padl + k;
    if (p >= 0 && p < L) acc += X[((size_t)b * L + p) * C + c] * W[c * Kw + k];
  }
  Y[i] = acc;
}

// mamba: causal depthwise conv (k=4, left pad 3) reading xs inside the xz
// buffer (row stride ldx), fused SiLU
__global__ void k_causal_dwconv_silu(const float* __restrict__ XZ, int ldx,
                                     const float* __restrict__ cw,
                                     const float* __restrict__ cb,
                                     float* __restrict__ U,
                                     int B, int L, int d_in)
{
  size_t i = (size_t)blockIdx.x * 256 + threadIdx.x;
  size_t total = (size_t)B * L * d_in;
  if (i >= total) return;
  int d = (int)(i % d_in);
  int l = (int)((i / d_in) % L);
  int b = (int)(i / ((size_t)d_in * L));
  float acc = cb[d];
  for (int k = 0; k < 4; ++k) {
    int p = l - 3 + k;
    if (p >= 0) acc += XZ[((size_t)b * L + p) * ldx + d] * cw[d * 4 + k];
  }
  U[i] = silu_f(acc);
}

__global__ void k_gate(const float* __restrict__ Yv, const float* __restrict__ Z,
                       int ldz, float* __restrict__ G, size_t M, int d_in)
{
  size_t i = (size_t)blockIdx.x * 256 + threadIdx.x;
  if (i >= M * (size_t)d_in) return;
  size_t r = i / d_in;
  int c = (int)(i % d_in);
  G[i] = Yv[i] * silu_f(Z[r * ldz + c]);
}

__global__ void k_add(const float* __restrict__ A, const float* __restrict__ Bv,
                      float* __restrict__ C, size_t n)
{
  size_t i = (size_t)blockIdx.x * 256 + threadIdx.x;
  if (i < n) C[i] = A[i] + Bv[i];
}

__global__ void k_mean_L(const float* __restrict__ X, float* __restrict__ Wm,
                         int B, int L, int C)
{
  int i = blockIdx.x * 256 + threadIdx.x;
  if (i >= B * C) return;
  int b = i / C, c = i % C;
  float s = 0.f;
  for (int l = 0; l < L; ++l) s += X[((size_t)b * L + l) * C + c];
  Wm[i] = s / (float)L;
}

template<int ACT>
__global__ void k_small_mm(const float* __restrict__ X, const float* __restrict__ W,
                           const float* __restrict__ bias, float* __restrict__ Y,
                           int B, int N, int K)
{
  int i = blockIdx.x * 256 + threadIdx.x;
  if (i >= B * N) return;
  int o = i % N, b = i / N;
  float acc = bias[o];
  for (int k = 0; k < K; ++k) acc += W[(size_t)o * K + k] * X[(size_t)b * K + k];
  Y[i] = apply_act(acc, ACT);
}

__global__ void k_attn_apply(float* __restrict__ X, const float* __restrict__ Wm,
                             int B, int L, int C)
{
  size_t i = (size_t)blockIdx.x * 256 + threadIdx.x;
  if (i >= (size_t)B * L * C) return;
  int c = (int)(i % C);
  int b = (int)(i / ((size_t)L * C));
  X[i] *= (Wm[b * C + c] + 1.f);   // x*w + x
}

// ---------------------------------------------------------------------------
// Host-side parameter structs / helpers
// ---------------------------------------------------------------------------
struct BNP   { const float *g, *b, *m, *v; };
struct DWSP  { const float *dw_w, *dw_b, *pw_w, *pw_b; };
struct MambaP{ const float *in_w, *conv_w, *conv_b, *x_w, *dt_w, *dt_b, *A_log, *Dv, *out_w; };
struct FFP   { DWSP ds; const float *pw_w, *pw_b; const float *aw1, *ab1, *aw2, *ab2; };
struct LayerP{ const float *ln1g, *ln1b; MambaP m; const float *ln2g, *ln2b; FFP ff; };
struct DSCP  { BNP bn1; DWSP c1; BNP bn2; DWSP c2; };

static inline dim3 g1d(size_t n) { return dim3((unsigned)((n + 255) / 256)); }

static void gemm_launch(hipStream_t s, const float* X, int lda, const float* W,
                        const float* bias, float* Y, int ldy,
                        int M, int N, int K, int act)
{
  dim3 gr((N + 63) / 64, (M + 31) / 32), bl(128);
  switch (act) {
    case 1: k_gemm_wmma<1><<<gr, bl, 0, s>>>(X, lda, W, bias, Y, ldy, M, N, K); break;
    case 2: k_gemm_wmma<2><<<gr, bl, 0, s>>>(X, lda, W, bias, Y, ldy, M, N, K); break;
    case 3: k_gemm_wmma<3><<<gr, bl, 0, s>>>(X, lda, W, bias, Y, ldy, M, N, K); break;
    case 4: k_gemm_wmma<4><<<gr, bl, 0, s>>>(X, lda, W, bias, Y, ldy, M, N, K); break;
    default:k_gemm_wmma<0><<<gr, bl, 0, s>>>(X, lda, W, bias, Y, ldy, M, N, K); break;
  }
}

// ---------------------------------------------------------------------------
// kernel_launch
// ---------------------------------------------------------------------------
extern "C" void kernel_launch(void* const* d_in, const int* in_sizes, int n_in,
                              void* d_out, int out_size, void* d_ws, size_t ws_size,
                              hipStream_t stream)
{
  (void)in_sizes; (void)n_in; (void)out_size; (void)ws_size;

  int pi = 0;
  auto P = [&]() -> const float* { return (const float*)d_in[pi++]; };

  const float* x = P();
  const float* stem_w = P(); const float* stem_b = P();

  auto rdBN    = [&]{ BNP r; r.g = P(); r.b = P(); r.m = P(); r.v = P(); return r; };
  auto rdDWS   = [&]{ DWSP r; r.dw_w = P(); r.dw_b = P(); r.pw_w = P(); r.pw_b = P(); return r; };
  auto rdMamba = [&]{ MambaP r; r.in_w = P(); r.conv_w = P(); r.conv_b = P(); r.x_w = P();
                      r.dt_w = P(); r.dt_b = P(); r.A_log = P(); r.Dv = P(); r.out_w = P(); return r; };
  auto rdFF    = [&](bool attn){ FFP r{}; r.ds = rdDWS(); r.pw_w = P(); r.pw_b = P();
                      if (attn) { r.aw1 = P(); r.ab1 = P(); r.aw2 = P(); r.ab2 = P(); } return r; };
  auto rdLayer = [&](bool attn){ LayerP r; r.ln1g = P(); r.ln1b = P(); r.m = rdMamba();
                      r.ln2g = P(); r.ln2b = P(); r.ff = rdFF(attn); return r; };
  auto rdDSC   = [&]{ DSCP r; r.bn1 = rdBN(); r.c1 = rdDWS(); r.bn2 = rdBN(); r.c2 = rdDWS(); return r; };

  LayerP mnet0[3]; for (int i = 0; i < 3; ++i) mnet0[i] = rdLayer(false);
  const float* c1w = P(); const float* c1b = P();
  LayerP mnet1[3]; for (int i = 0; i < 3; ++i) mnet1[i] = rdLayer(false);
  const float* c2w = P(); const float* c2b = P();
  LayerP mnet2[3]; for (int i = 0; i < 3; ++i) mnet2[i] = rdLayer(false);
  const float* c3w = P(); const float* c3b = P();
  LayerP mnet3[3]; for (int i = 0; i < 3; ++i) mnet3[i] = rdLayer(true);
  DSCP dsc3 = rdDSC(); DSCP dsc5 = rdDSC(); DSCP dsc7 = rdDSC();

  // --- workspace bump allocator: 16B-aligned, >=1KB slack after each buffer
  //     (GEMM tail loads may over-read past a row's logical end) ---
  float* wsf = (float*)d_ws;
  size_t off = 0;
  auto alloc = [&](size_t n) { float* p = wsf + off; off += ((n + 255) & ~(size_t)255) + 256; return p; };

  const int B = 8;
  float* nch0 = alloc((size_t)B * 32 * 2048);
  float* nch1 = alloc((size_t)B * 32 * 2048);
  float* blc  = alloc((size_t)B * 2048 * 16);   // max B*L*C over all mnets
  float* h2a  = alloc((size_t)B * 128 * 128);
  float* fsum = alloc((size_t)B * 128 * 128);

  // --- mnet runner (h is (B,L,D) channel-last, updated in place) ---
  auto run_mnet = [&](float* h, int L, int D, bool attn, const LayerP* Ls) {
    const size_t M = (size_t)B * L;
    const int d_in = 2 * D, N2 = 2 * D;
    int r = (D + 15) / 16; if (r < 1) r = 1;
    const int rp  = (r + 3) & ~3;                // padded row stride for dtr
    const int hid = D / 2;
    size_t mark = off;
    float* hn  = alloc(M * D);
    float* xz  = alloc(M * (size_t)(2 * d_in));
    float* u   = alloc(M * d_in);
    float* dtr = alloc(M * (size_t)rp);          // dbl[:, :r]
    float* bc  = alloc(M * (size_t)(2 * N2));    // dbl[:, r:]  (B | C), aligned
    float* dt  = alloc(M * d_in);
    float* yv  = alloc(M * d_in);
    float* gbuf= alloc(M * d_in);
    float* t1  = alloc(M * D);
    float* t2  = alloc(M * hid);
    float* wm0 = alloc((size_t)B * hid);
    float* wm1 = alloc((size_t)B * hid);
    float* wm2 = alloc((size_t)B * hid);

    for (int l = 0; l < 3; ++l) {
      const LayerP& lp = Ls[l];
      // hn = LN(h)
      k_layernorm<<<dim3((unsigned)M), 32, 0, stream>>>(h, lp.ln1g, lp.ln1b, hn, (int)M, D);
      // xz = hn @ in_w^T
      gemm_launch(stream, hn, D, lp.m.in_w, nullptr, xz, 2 * d_in, (int)M, 2 * d_in, D, 0);
      // u = silu(causal dwconv(xs))
      k_causal_dwconv_silu<<<g1d(M * d_in), 256, 0, stream>>>(xz, 2 * d_in, lp.m.conv_w,
                                                              lp.m.conv_b, u, B, L, d_in);
      // x_w projection split so B/C land 16B-aligned:
      //   dtr = u @ x_w[0:r]^T ;  bc = u @ x_w[r:]^T  (B at col 0, C at col N2)
      gemm_launch(stream, u, d_in, lp.m.x_w, nullptr, dtr, rp, (int)M, r, d_in, 0);
      gemm_launch(stream, u, d_in, lp.m.x_w + (size_t)r * d_in, nullptr, bc, 2 * N2,
                  (int)M, 2 * N2, d_in, 0);
      // dt = softplus(dtr @ dt_w^T + dt_b)
      gemm_launch(stream, dtr, rp, lp.m.dt_w, lp.m.dt_b, dt, d_in, (int)M, d_in, r, 4);
      // selective scan
      {
        dim3 sg(d_in / 32, B);
        int nper = N2 / 8;
        if      (nper == 4 ) k_scan<4 ><<<sg, 256, 0, stream>>>(dt, bc, bc + N2, 2 * N2, u, lp.m.A_log, lp.m.Dv, yv, L, d_in, N2);
        else if (nper == 8 ) k_scan<8 ><<<sg, 256, 0, stream>>>(dt, bc, bc + N2, 2 * N2, u, lp.m.A_log, lp.m.Dv, yv, L, d_in, N2);
        else if (nper == 16) k_scan<16><<<sg, 256, 0, stream>>>(dt, bc, bc + N2, 2 * N2, u, lp.m.A_log, lp.m.Dv, yv, L, d_in, N2);
        else                 k_scan<32><<<sg, 256, 0, stream>>>(dt, bc, bc + N2, 2 * N2, u, lp.m.A_log, lp.m.Dv, yv, L, d_in, N2);
      }
      // g = y * silu(z);  mamba_out = g @ out_w^T;  h = mamba_out + hn
      k_gate<<<g1d(M * d_in), 256, 0, stream>>>(yv, xz + d_in, 2 * d_in, gbuf, M, d_in);
      gemm_launch(stream, gbuf, d_in, lp.m.out_w, nullptr, t1, D, (int)M, D, d_in, 0);
      k_add<<<g1d(M * D), 256, 0, stream>>>(t1, hn, h, M * D);

      // hn = LN(h);  feedforward
      k_layernorm<<<dim3((unsigned)M), 32, 0, stream>>>(h, lp.ln2g, lp.ln2b, hn, (int)M, D);
      const int kk = attn ? 3 : 5;
      k_dwconv_nlc<<<g1d(M * D), 256, 0, stream>>>(hn, lp.ff.ds.dw_w, lp.ff.ds.dw_b, t1,
                                                   B, L, D, kk, (kk - 1) / 2);
      gemm_launch(stream, t1, D, lp.ff.ds.pw_w, lp.ff.ds.pw_b, t2, hid, (int)M, hid, D, 1);
      if (attn) {
        k_mean_L<<<g1d((size_t)B * hid), 256, 0, stream>>>(t2, wm0, B, L, hid);
        k_small_mm<1><<<g1d((size_t)B * (hid / 2)), 256, 0, stream>>>(wm0, lp.ff.aw1, lp.ff.ab1, wm1, B, hid / 2, hid);
        k_small_mm<3><<<g1d((size_t)B * hid),       256, 0, stream>>>(wm1, lp.ff.aw2, lp.ff.ab2, wm2, B, hid, hid / 2);
        k_attn_apply<<<g1d(M * hid), 256, 0, stream>>>(t2, wm2, B, L, hid);
      }
      gemm_launch(stream, t2, hid, lp.ff.pw_w, lp.ff.pw_b, t1, D, (int)M, D, hid, 0);
      k_add<<<g1d(M * D), 256, 0, stream>>>(t1, hn, h, M * D);
    }
    off = mark;
  };

  // --- dsconv block runner, channel-last (B,L,C) ---
  auto run_dsc = [&](const float* xin, float* xout, const DSCP& p, int kk, int L, int C) {
    const size_t M = (size_t)B * L;
    const int hid = C / 2;
    size_t mark = off;
    float* b1 = alloc(M * C);
    float* d1 = alloc(M * C);
    float* h1 = alloc(M * hid);
    float* b2 = alloc(M * hid);
    float* d2 = alloc(M * hid);
    float* ho = alloc(M * C);
    k_bn_gelu<<<g1d(M * C), 256, 0, stream>>>(xin, p.bn1.g, p.bn1.b, p.bn1.m, p.bn1.v, b1, M * C, C);
    k_dwconv_nlc<<<g1d(M * C), 256, 0, stream>>>(b1, p.c1.dw_w, p.c1.dw_b, d1, B, L, C, kk, (kk - 1) / 2);
    gemm_launch(stream, d1, C, p.c1.pw_w, p.c1.pw_b, h1, hid, (int)M, hid, C, 0);
    k_bn_gelu<<<g1d(M * hid), 256, 0, stream>>>(h1, p.bn2.g, p.bn2.b, p.bn2.m, p.bn2.v, b2, M * hid, hid);
    k_dwconv_nlc<<<g1d(M * hid), 256, 0, stream>>>(b2, p.c2.dw_w, p.c2.dw_b, d2, B, L, hid, kk, (kk - 1) / 2);
    gemm_launch(stream, d2, hid, p.c2.pw_w, p.c2.pw_b, ho, C, (int)M, C, hid, 0);
    k_add<<<g1d(M * C), 256, 0, stream>>>(xin, ho, xout, M * C);
    off = mark;
  };

  // =============================== pipeline ===============================
  // stem: (8,3,16384) -> (8,16,2048)
  k_conv1d_nch<<<g1d((size_t)B * 16 * 2048), 256, 0, stream>>>(x, stem_w, stem_b, nch0,
                                                               B, 3, 16384, 16, 2048, 11, 8, 5);
  // mnet0 @ (16, 2048)
  k_nch2nlc<<<g1d((size_t)B * 16 * 2048), 256, 0, stream>>>(nch0, blc, B, 16, 2048);
  run_mnet(blc, 2048, 16, false, mnet0);
  k_nlc2nch<<<g1d((size_t)B * 16 * 2048), 256, 0, stream>>>(blc, nch0, B, 16, 2048);
  // c1 (16->32, k7 p3) + pool4
  k_conv1d_nch<<<g1d((size_t)B * 32 * 2048), 256, 0, stream>>>(nch0, c1w, c1b, nch1,
                                                               B, 16, 2048, 32, 2048, 7, 1, 3);
  k_maxpool_nch<<<g1d((size_t)B * 32 * 512), 256, 0, stream>>>(nch1, nch0, B, 32, 2048, 4);
  // mnet1 @ (32, 512)
  k_nch2nlc<<<g1d((size_t)B * 32 * 512), 256, 0, stream>>>(nch0, blc, B, 32, 512);
  run_mnet(blc, 512, 32, false, mnet1);
  k_nlc2nch<<<g1d((size_t)B * 32 * 512), 256, 0, stream>>>(blc, nch0, B, 32, 512);
  // c2 (32->64, k5 p2) + pool2
  k_conv1d_nch<<<g1d((size_t)B * 64 * 512), 256, 0, stream>>>(nch0, c2w, c2b, nch1,
                                                              B, 32, 512, 64, 512, 5, 1, 2);
  k_maxpool_nch<<<g1d((size_t)B * 64 * 256), 256, 0, stream>>>(nch1, nch0, B, 64, 512, 2);
  // mnet2 @ (64, 256)
  k_nch2nlc<<<g1d((size_t)B * 64 * 256), 256, 0, stream>>>(nch0, blc, B, 64, 256);
  run_mnet(blc, 256, 64, false, mnet2);
  k_nlc2nch<<<g1d((size_t)B * 64 * 256), 256, 0, stream>>>(blc, nch0, B, 64, 256);
  // c3 (64->128, k3 p1) + pool2
  k_conv1d_nch<<<g1d((size_t)B * 128 * 256), 256, 0, stream>>>(nch0, c3w, c3b, nch1,
                                                               B, 64, 256, 128, 256, 3, 1, 1);
  k_maxpool_nch<<<g1d((size_t)B * 128 * 128), 256, 0, stream>>>(nch1, nch0, B, 128, 256, 2);
  // mnet3 @ (128, 128) with attention; stay channel-last afterwards
  k_nch2nlc<<<g1d((size_t)B * 128 * 128), 256, 0, stream>>>(nch0, blc, B, 128, 128);
  run_mnet(blc, 128, 128, true, mnet3);
  // mconv: three dsconv blocks in channel-last
  run_dsc(blc, h2a, dsc3, 3, 128, 128);
  run_dsc(h2a, h2a, dsc5, 5, 128, 128);
  run_dsc(h2a, h2a, dsc7, 7, 128, 128);
  // out = h + h2, back to (B,C,L)
  k_add<<<g1d((size_t)B * 128 * 128), 256, 0, stream>>>(blc, h2a, fsum, (size_t)B * 128 * 128);
  k_nlc2nch<<<g1d((size_t)B * 128 * 128), 256, 0, stream>>>(fsum, (float*)d_out, B, 128, 128);
}
